// HeteroRGCN_31610959298704
// MI455X (gfx1250) — matrix operand order repaired and checked by reading, not compiled, not run on hardware
//
#include <hip/hip_runtime.h>

typedef __attribute__((ext_vector_type(16))) _Float16 v16h;
typedef __attribute__((ext_vector_type(8)))  float    v8f;

#define NU   100000
#define NI   100000
#define NE   1000000
#define IND  256
#define HID  128
#define OUTD 16

// ---------------------------------------------------------------------------
// Weight pre-pack: W[K,N] (f32, row-major) -> Wp (f16, WMMA-fragment-major).
// Fragment element t = ((w*NCH + c)*32 + L)*16 + j  holds
//   W[k, col] with k = c*32 + (L>>4)*8 + j + (j>=8 ? 8 : 0), col = w*16 + (L&15)
// so a wave's B fragment for (N-tile w, chunk c) is 32 consecutive v16h,
// one per lane: a single coalesced 32-byte load per lane, no cvt in the loop.
// ---------------------------------------------------------------------------
__global__ void pack_w(const float* __restrict__ W, _Float16* __restrict__ Wp,
                       int K, int N) {
  const int t = blockIdx.x * blockDim.x + threadIdx.x;
  if (t >= K * N) return;
  const int nch = K >> 5;
  int idx = t;
  const int j = idx & 15;  idx >>= 4;
  const int L = idx & 31;  idx >>= 5;
  const int c = idx % nch;
  const int w = idx / nch;
  const int k   = c * 32 + (L >> 4) * 8 + j + ((j < 8) ? 0 : 8);
  const int col = w * 16 + (L & 15);
  Wp[t] = (_Float16)W[(size_t)k * N + col];
}

// ---------------------------------------------------------------------------
// Wide-N GEMM: out[M,128] = A[M,K] @ W[K,128] + bias, K in {128,256}.
// Block = 8 waves = one 16-row M strip; wave w owns N-tile w.
// A strip staged in LDS in fragment-major f16 order (cvt done once);
// B fragments come pre-packed from global. Inner loop: 1 LDS load +
// 1 global v16h load + 1 v_wmma per chunk.
// ---------------------------------------------------------------------------
template <int K>
__global__ __launch_bounds__(256) void gemm128(const float* __restrict__ A,
                                               const _Float16* __restrict__ Wp,
                                               const float* __restrict__ bias,
                                               float* __restrict__ out) {
  constexpr int N   = 128;
  constexpr int NCH = K / 32;
  __shared__ alignas(32) _Float16 sA[16 * K];   // [chunk][lane][16] fragment-major

  const int tileM = blockIdx.x;

  // Cooperative stage: element idx = (c*32 + L)*16 + j  holds
  //   A[tileM*16 + (L&15), c*32 + (L>>4)*8 + j + (j>=8 ? 8 : 0)]  as f16
  for (int idx = threadIdx.x; idx < 16 * K; idx += 256) {
    const int j    = idx & 15;
    const int L    = (idx >> 4) & 31;
    const int c    = idx >> 9;
    const int row  = L & 15;
    const int half = L >> 4;
    const int kk   = c * 32 + half * 8 + j + ((j < 8) ? 0 : 8);
    sA[idx] = (_Float16)A[(size_t)(tileM * 16 + row) * K + kk];
  }
  __syncthreads();

  const int wave = threadIdx.x >> 5;            // N tile (0..7)
  const int lane = threadIdx.x & 31;
  const int ln   = lane & 15;
  const int half = lane >> 4;
  const int col  = wave * 16 + ln;

  const v16h* sAv = (const v16h*)sA;
  const v16h* bp  = (const v16h*)Wp + (size_t)wave * NCH * 32 + lane;

  v8f acc = {};
#pragma unroll
  for (int c = 0; c < NCH; ++c) {
    const v16h a = sAv[c * 32 + lane];
    const v16h b = bp[c * 32];
    acc = __builtin_amdgcn_wmma_f32_16x16x32_f16(false, a, false, b,
                                                 (short)0, acc, false, false);
  }

  const float bv = bias[col];
#pragma unroll
  for (int r = 0; r < 8; ++r) {
    const int rowOut = tileM * 16 + half * 8 + r;     // C layout: M = half*8 + r
    out[(size_t)rowOut * N + col] = acc[r] + bv;
  }
}

// ---------------------------------------------------------------------------
// Narrow-N GEMM: out[M,16] = A[M,128] @ W[128,16] + bias. One wave per M tile.
// ---------------------------------------------------------------------------
__global__ __launch_bounds__(256) void gemm_out16(const float* __restrict__ A,
                                                  const _Float16* __restrict__ Wp,
                                                  const float* __restrict__ bias,
                                                  float* __restrict__ out, int M) {
  constexpr int K = HID, N = OUTD, NCH = K / 32;
  const int wave  = threadIdx.x >> 5;
  const int lane  = threadIdx.x & 31;
  const int tileM = blockIdx.x * 8 + wave;
  if (tileM * 16 >= M) return;                 // wave-uniform guard (EXEC stays full)

  const int ln   = lane & 15;
  const int half = lane >> 4;
  const int row  = tileM * 16 + ln;

  v8f acc = {};
#pragma unroll
  for (int c = 0; c < NCH; ++c) {
    v16h a;
    const float* ap = A + (size_t)row * K + c * 32 + half * 8;
#pragma unroll
    for (int j = 0; j < 8; ++j) {
      a[j]     = (_Float16)ap[j];
      a[8 + j] = (_Float16)ap[16 + j];
    }
    const v16h b = ((const v16h*)Wp)[c * 32 + lane];
    acc = __builtin_amdgcn_wmma_f32_16x16x32_f16(false, a, false, b,
                                                 (short)0, acc, false, false);
  }

  const float bv = bias[ln];
#pragma unroll
  for (int r = 0; r < 8; ++r)
    out[(size_t)(tileM * 16 + half * 8 + r) * N + ln] = acc[r] + bv;
}

// ---------------------------------------------------------------------------
// Edge scatter: one wave per edge; each lane moves 4 floats of the 128-wide
// row. sums[dst] += wh[src]; cnt[dst] += 1.  (segment-sum for the mean)
// ---------------------------------------------------------------------------
__global__ __launch_bounds__(256) void scatter_edges(const float* __restrict__ wh,
                                                     const int* __restrict__ src,
                                                     const int* __restrict__ dst,
                                                     float* __restrict__ sums,
                                                     float* __restrict__ cnt,
                                                     int nEdge) {
  const int t    = blockIdx.x * blockDim.x + threadIdx.x;
  const int edge = t >> 5;
  if (edge >= nEdge) return;
  const int lane = t & 31;
  const int s = src[edge];
  const int d = dst[edge];

  const float4 v = *(const float4*)(wh + (size_t)s * HID + lane * 4);
  float* o = sums + (size_t)d * HID + lane * 4;
  atomicAdd(o + 0, v.x);
  atomicAdd(o + 1, v.y);
  atomicAdd(o + 2, v.z);
  atomicAdd(o + 3, v.w);
  if (lane == 0) atomicAdd(cnt + d, 1.0f);
}

__global__ void zero_f32(float* __restrict__ p, int n) {
  const int t = blockIdx.x * blockDim.x + threadIdx.x;
  if (t < n) p[t] = 0.0f;
}

// mean (zero-degree -> 0) + optional leaky_relu, in place
__global__ void seg_finish(float* __restrict__ h, const float* __restrict__ cnt,
                           int nNodes, int leaky) {
  const int t = blockIdx.x * blockDim.x + threadIdx.x;
  if (t >= nNodes * HID) return;
  const int   node = t >> 7;                   // HID == 128
  const float c    = cnt[node];
  float v = (c > 0.0f) ? (h[t] / c) : 0.0f;
  if (leaky) v = (v > 0.0f) ? v : 0.01f * v;
  h[t] = v;
}

// ---------------------------------------------------------------------------
// Live dataflow (dead code eliminated — only the user-output chain matters):
//   P0 = feat_user @ Win_user + bin_user
//   P1 = P0 @ W0_clicks + b0_clicks                 (wh_u)
//   P2 = leaky(seg_mean_clicks(P1))                 (h_i, layer 0)
//   P0 = P2 @ W1_cb + b1_cb                         (wh_i)
//   P1 = seg_mean_cb(P0)                            (h_u, layer 1)
//   out = P1 @ Wout + bout
// ---------------------------------------------------------------------------
extern "C" void kernel_launch(void* const* d_in, const int* in_sizes, int n_in,
                              void* d_out, int out_size, void* d_ws, size_t ws_size,
                              hipStream_t stream) {
  (void)in_sizes; (void)n_in; (void)out_size; (void)ws_size;

  const float* feat_user  = (const float*)d_in[0];
  const int*   src_clicks = (const int*)d_in[2];
  const int*   dst_clicks = (const int*)d_in[3];
  const int*   src_cb     = (const int*)d_in[4];
  const int*   dst_cb     = (const int*)d_in[5];
  const float* Win_user   = (const float*)d_in[6];
  const float* bin_user   = (const float*)d_in[7];
  const float* W0c        = (const float*)d_in[10];
  const float* b0c        = (const float*)d_in[11];
  const float* W1cb       = (const float*)d_in[16];
  const float* b1cb       = (const float*)d_in[17];
  const float* Wout       = (const float*)d_in[18];
  const float* bout       = (const float*)d_in[19];

  float* P0   = (float*)d_ws;                  // 100000*128 floats each
  float* P1   = P0 + (size_t)NU * HID;
  float* P2   = P1 + (size_t)NU * HID;
  float* cntI = P2 + (size_t)NI * HID;
  float* cntU = cntI + NI;
  _Float16* WpIn  = (_Float16*)(cntU + NU);    // packed f16 weights (L2-resident)
  _Float16* Wp0c  = WpIn  + IND * HID;
  _Float16* Wp1cb = Wp0c  + HID * HID;
  _Float16* WpOut = Wp1cb + HID * HID;

  const int nodeTiles = NU / 16;               // 6250, exact
  const int nz        = NU * HID;

  // 0) pack weights once per launch (tiny)
  pack_w<<<(IND * HID  + 255) / 256, 256, 0, stream>>>(Win_user, WpIn,  IND, HID);
  pack_w<<<(HID * HID  + 255) / 256, 256, 0, stream>>>(W0c,      Wp0c,  HID, HID);
  pack_w<<<(HID * HID  + 255) / 256, 256, 0, stream>>>(W1cb,     Wp1cb, HID, HID);
  pack_w<<<(HID * OUTD + 255) / 256, 256, 0, stream>>>(Wout,     WpOut, HID, OUTD);

  // 1) input projection (K=256)
  gemm128<IND><<<nodeTiles, 256, 0, stream>>>(feat_user, WpIn, bin_user, P0);
  // 2) layer-0 clicks linear (K=128)
  gemm128<HID><<<nodeTiles, 256, 0, stream>>>(P0, Wp0c, b0c, P1);
  // 3) seg-mean over clicks into items
  zero_f32<<<(nz + 255) / 256, 256, 0, stream>>>(P2, nz);
  zero_f32<<<(NI + 255) / 256, 256, 0, stream>>>(cntI, NI);
  scatter_edges<<<(NE * 32) / 256, 256, 0, stream>>>(P1, src_clicks, dst_clicks, P2, cntI, NE);
  seg_finish<<<(nz + 255) / 256, 256, 0, stream>>>(P2, cntI, NI, 1);  // + leaky
  // 4) layer-1 clicked_by linear (K=128)
  gemm128<HID><<<nodeTiles, 256, 0, stream>>>(P2, Wp1cb, b1cb, P0);
  // 5) seg-mean over cb into users
  zero_f32<<<(nz + 255) / 256, 256, 0, stream>>>(P1, nz);
  zero_f32<<<(NU + 255) / 256, 256, 0, stream>>>(cntU, NU);
  scatter_edges<<<(NE * 32) / 256, 256, 0, stream>>>(P0, src_cb, dst_cb, P1, cntU, NE);
  seg_finish<<<(nz + 255) / 256, 256, 0, stream>>>(P1, cntU, NU, 0);  // no leaky
  // 6) output projection
  gemm_out16<<<(nodeTiles + 7) / 8, 256, 0, stream>>>(P1, WpOut, bout, (float*)d_out, NU);
}
